// ChessNNUE_42820823941279
// MI455X (gfx1250) — matrix operand order repaired and checked by reading, not compiled, not run on hardware
//
#include <hip/hip_runtime.h>
#include <hip/hip_bf16.h>
#include <cmath>

// ---------------- problem constants (match reference) ----------------
#define B_SZ 4096
#define FEAT 40960
#define HID  1024
#define L1N  64
#define L2N  32

typedef __attribute__((ext_vector_type(4)))  float    v4f;
typedef __attribute__((ext_vector_type(8)))  float    v8f;
typedef __attribute__((ext_vector_type(8)))  _Float16 v8h;
typedef __attribute__((ext_vector_type(16))) _Float16 v16h;

// ---------------------------------------------------------------------
// Kernel 1: transpose ft_w [HID, FEAT] -> ft_wT [FEAT, HID] (f32).
// Makes the per-feature gather in the FT kernel fully coalesced
// (one contiguous 4KB row per active feature). ft_wT (167MB) lives in
// the 192MB L2 for the duration of the dispatch.
// ---------------------------------------------------------------------
__global__ void __launch_bounds__(256) nnue_transpose_ftw(
    const float* __restrict__ src, float* __restrict__ dst) {
  __shared__ float tile[32][33];
  const int f0 = blockIdx.x * 32, h0 = blockIdx.y * 32;
  const int tx = threadIdx.x, ty = threadIdx.y;
#pragma unroll
  for (int i = 0; i < 32; i += 8)
    tile[ty + i][tx] = src[(size_t)(h0 + ty + i) * FEAT + (f0 + tx)];
  __syncthreads();
#pragma unroll
  for (int i = 0; i < 32; i += 8)
    dst[(size_t)(f0 + ty + i) * HID + (h0 + tx)] = tile[tx][ty + i];
}

// ---------------------------------------------------------------------
// Kernel 2: convert l1_w / l2_w to f16 for WMMA.
// ---------------------------------------------------------------------
__global__ void nnue_cvt_weights(const float* __restrict__ l1w,
                                 const float* __restrict__ l2w,
                                 _Float16* __restrict__ l1w16,
                                 _Float16* __restrict__ l2w16) {
  const int i = blockIdx.x * blockDim.x + threadIdx.x;
  const int n1 = L1N * 2 * HID;
  if (i < n1) l1w16[i] = (_Float16)l1w[i];
  const int j = i - n1;
  if (j >= 0 && j < L2N * L1N) l2w16[j] = (_Float16)l2w[j];
}

// ---------------------------------------------------------------------
// Kernel 3: sparse feature transformer.
// One workgroup (256 thr) per position: NT-stream both 160KB feature
// rows, collect nonzero indices into LDS, then accumulate the ~30
// matching rows of ft_wT (L2-resident) + bias, apply the stm-based
// perspective concat + clip, store h1 as f16 (WMMA A operand food).
// ---------------------------------------------------------------------
#define MAXF 256
__global__ void __launch_bounds__(256) nnue_ft(
    const float* __restrict__ wf, const float* __restrict__ bf,
    const float* __restrict__ stm, const float* __restrict__ ftwT,
    const float* __restrict__ ftb, _Float16* __restrict__ h1) {
  __shared__ int s_cnt[2];
  __shared__ int s_idx[2][MAXF];
  const int b = blockIdx.x, tid = threadIdx.x;
  if (tid < 2) s_cnt[tid] = 0;
  __syncthreads();

  const float* rows[2] = {wf + (size_t)b * FEAT, bf + (size_t)b * FEAT};
  for (int p = 0; p < 2; ++p) {
    const v4f* r = (const v4f*)rows[p];
    for (int i = tid; i < FEAT / 4; i += 256) {
      v4f v = __builtin_nontemporal_load(r + i);  // streaming: don't pollute L2
#pragma unroll
      for (int c = 0; c < 4; ++c) {
        if (v[c] != 0.0f) {
          int k = atomicAdd(&s_cnt[p], 1);
          if (k < MAXF) s_idx[p][k] = 4 * i + c;
        }
      }
    }
  }
  __syncthreads();

  float accW[4], accB[4];
#pragma unroll
  for (int j = 0; j < 4; ++j) {
    float bb = ftb[tid + 256 * j];
    accW[j] = bb; accB[j] = bb;
  }
  const int cw = min(s_cnt[0], MAXF), cb = min(s_cnt[1], MAXF);
  for (int k = 0; k < cw; ++k) {
    const float* col = ftwT + (size_t)s_idx[0][k] * HID;
#pragma unroll
    for (int j = 0; j < 4; ++j) accW[j] += col[tid + 256 * j];
  }
  for (int k = 0; k < cb; ++k) {
    const float* col = ftwT + (size_t)s_idx[1][k] * HID;
#pragma unroll
    for (int j = 0; j < 4; ++j) accB[j] += col[tid + 256 * j];
  }

  const float st = stm[b];
  _Float16* o = h1 + (size_t)b * (2 * HID);
#pragma unroll
  for (int j = 0; j < 4; ++j) {
    float fw = st * accW[j] + (1.0f - st) * accB[j];
    float sb = st * accB[j] + (1.0f - st) * accW[j];
    o[tid + 256 * j]       = (_Float16)fminf(fmaxf(fw, 0.0f), 1.0f);
    o[HID + tid + 256 * j] = (_Float16)fminf(fmaxf(sb, 0.0f), 1.0f);
  }
}

// ---------------------------------------------------------------------
// WMMA operand loaders (CDNA5 16-bit layouts, wave32).
// A (MxK=16x32): lane<16 -> row m0+lane, K {k0..k0+7, k0+16..k0+23};
//                lane>=16 -> row m0+lane-16, K {k0+8..15, k0+24..31}.
// B (KxN=32x16) from row-major W[N][K]: lane half h, col n0+(lane&15),
//                K {k0+8h..k0+8h+7, k0+16+8h..k0+16+8h+7} (same as A).
// ---------------------------------------------------------------------
__device__ inline v16h load_a16(const _Float16* base, int ld, int m0, int k0,
                                int lane) {
  const int h = lane >> 4;
  const _Float16* p = base + (size_t)(m0 + (lane & 15)) * ld + k0 + 8 * h;
  v8h lo = *(const v8h*)p;
  v8h hi = *(const v8h*)(p + 16);
  return __builtin_shufflevector(lo, hi, 0, 1, 2, 3, 4, 5, 6, 7, 8, 9, 10, 11,
                                 12, 13, 14, 15);
}
__device__ inline v16h load_b16(const _Float16* base, int ld, int n0, int k0,
                                int lane) {
  const int h = lane >> 4;
  const _Float16* p = base + (size_t)(n0 + (lane & 15)) * ld + k0 + 8 * h;
  v8h lo = *(const v8h*)p;
  v8h hi = *(const v8h*)(p + 16);
  return __builtin_shufflevector(lo, hi, 0, 1, 2, 3, 4, 5, 6, 7, 8, 9, 10, 11,
                                 12, 13, 14, 15);
}

// ---------------------------------------------------------------------
// Kernel 4: MLP head, WMMA f32_16x16x32_f16.
// 8 waves/block, each wave owns a 16-row M tile:
//   GEMM1 [16x2048]@[2048x64] -> 4 accum tiles (64 K-steps)
//   bias+clip -> LDS relayout -> GEMM2 [16x64]@[64x32] (4 WMMAs)
//   bias+clip -> LDS -> lanes 0..15 do the 32-dot + sigmoid.
// ---------------------------------------------------------------------
__global__ void __launch_bounds__(256) nnue_mlp(
    const _Float16* __restrict__ h1, const _Float16* __restrict__ l1w16,
    const float* __restrict__ l1b, const _Float16* __restrict__ l2w16,
    const float* __restrict__ l2b, const float* __restrict__ l3w,
    const float* __restrict__ l3b, float* __restrict__ out) {
  __shared__ __align__(16) _Float16 s_h2[8][16 * 64];
  __shared__ __align__(16) float    s_h3[8][16 * 32];
  const int tid = threadIdx.x;
  const int wave = tid >> 5, lane = tid & 31;
  const int m0 = blockIdx.x * 128 + wave * 16;
  const int half = lane >> 4, nl = lane & 15;

  // ---- GEMM1: h1[16 x 2048] x l1_w^T -> h2[16 x 64] ----
  v8f acc[4] = {};
  for (int k0 = 0; k0 < 2 * HID; k0 += 32) {
    v16h a = load_a16(h1, 2 * HID, m0, k0, lane);
#pragma unroll
    for (int nt = 0; nt < 4; ++nt) {
      v16h bm = load_b16(l1w16, 2 * HID, nt * 16, k0, lane);
      acc[nt] = __builtin_amdgcn_wmma_f32_16x16x32_f16(
          false, a, false, bm, (short)0, acc[nt], false, false);
    }
  }
  // bias + clip, re-layout through LDS into A-operand order (f16)
#pragma unroll
  for (int nt = 0; nt < 4; ++nt) {
    const float bias = l1b[nt * 16 + nl];
#pragma unroll
    for (int j = 0; j < 8; ++j) {
      float v = fminf(fmaxf(acc[nt][j] + bias, 0.0f), 1.0f);
      s_h2[wave][(j + 8 * half) * 64 + nt * 16 + nl] = (_Float16)v;
    }
  }
  __syncthreads();

  // ---- GEMM2: h2[16 x 64] x l2_w^T -> h3[16 x 32] ----
  v8f acc2[2] = {};
#pragma unroll
  for (int k0 = 0; k0 < L1N; k0 += 32) {
    v16h a = load_a16(&s_h2[wave][0], L1N, 0, k0, lane);
#pragma unroll
    for (int nt = 0; nt < 2; ++nt) {
      v16h bm = load_b16(l2w16, L1N, nt * 16, k0, lane);
      acc2[nt] = __builtin_amdgcn_wmma_f32_16x16x32_f16(
          false, a, false, bm, (short)0, acc2[nt], false, false);
    }
  }
#pragma unroll
  for (int nt = 0; nt < 2; ++nt) {
    const float bias = l2b[nt * 16 + nl];
#pragma unroll
    for (int j = 0; j < 8; ++j)
      s_h3[wave][(j + 8 * half) * 32 + nt * 16 + nl] =
          fminf(fmaxf(acc2[nt][j] + bias, 0.0f), 1.0f);
  }
  __syncthreads();

  // ---- layer 3 + sigmoid: one row per lane (lanes 0..15) ----
  if (lane < 16) {
    float s = l3b[0];
#pragma unroll
    for (int n = 0; n < L2N; ++n) s += s_h3[wave][lane * 32 + n] * l3w[n];
    const int row = m0 + lane;
    out[row] = 1.0f / (1.0f + expf(-s));  // sigmoid output first
    out[B_SZ + row] = s;                  // then raw
  }
}

// ---------------------------------------------------------------------
extern "C" void kernel_launch(void* const* d_in, const int* in_sizes, int n_in,
                              void* d_out, int out_size, void* d_ws,
                              size_t ws_size, hipStream_t stream) {
  (void)in_sizes; (void)n_in; (void)out_size; (void)ws_size;
  const float* wf  = (const float*)d_in[0];
  const float* bf  = (const float*)d_in[1];
  const float* stm = (const float*)d_in[2];
  const float* ftw = (const float*)d_in[3];
  const float* ftb = (const float*)d_in[4];
  const float* l1w = (const float*)d_in[5];
  const float* l1b = (const float*)d_in[6];
  const float* l2w = (const float*)d_in[7];
  const float* l2b = (const float*)d_in[8];
  const float* l3w = (const float*)d_in[9];
  const float* l3b = (const float*)d_in[10];

  // workspace layout (bytes)
  const size_t off_ftwT  = 0;                                   // 167,772,160
  const size_t off_h1    = off_ftwT + (size_t)FEAT * HID * 4;   //  16,777,216
  const size_t off_l1w16 = off_h1 + (size_t)B_SZ * 2 * HID * 2; //     262,144
  const size_t off_l2w16 = off_l1w16 + (size_t)L1N * 2 * HID * 2;
  char* ws = (char*)d_ws;
  float*    ftwT  = (float*)(ws + off_ftwT);
  _Float16* h1f   = (_Float16*)(ws + off_h1);
  _Float16* l1w16 = (_Float16*)(ws + off_l1w16);
  _Float16* l2w16 = (_Float16*)(ws + off_l2w16);

  nnue_transpose_ftw<<<dim3(FEAT / 32, HID / 32), dim3(32, 8), 0, stream>>>(
      ftw, ftwT);
  const int ncvt = L1N * 2 * HID + L2N * L1N;
  nnue_cvt_weights<<<(ncvt + 255) / 256, 256, 0, stream>>>(l1w, l2w, l1w16,
                                                           l2w16);
  nnue_ft<<<B_SZ, 256, 0, stream>>>(wf, bf, stm, ftwT, ftb, h1f);
  nnue_mlp<<<B_SZ / 128, 256, 0, stream>>>(h1f, l1w16, l1b, l2w16, l2b, l3w,
                                           l3b, (float*)d_out);
}
